// SparseMoeFFN_22436909154496
// MI455X (gfx1250) — matrix operand order repaired and checked by reading, not compiled, not run on hardware
//
#include <hip/hip_runtime.h>

// ---------------------------------------------------------------------------
// Sparse MoE SwiGLU FFN for MI455X (gfx1250): bf16 WMMA grouped GEMMs.
//   T=8192 tokens, H=1024, I=2048, E=8 experts, top-2, renormalized.
// Routed FLOPs ~206G -> compute-bound in bf16 WMMA; fp32 scalar is ~8x off
// the matrix path (16x16x4 vs 16x16x32). Weights converted f32->bf16 per call
// (memory-bound, unavoidable: no cross-call caching allowed).
// GEMM loops: register double-buffered global->LDS staging, one A fragment
// reused across 4 WMMAs per wave per K-step.
// ---------------------------------------------------------------------------

typedef __bf16 bf16;
typedef __attribute__((ext_vector_type(16))) __bf16 v16bf;
typedef __attribute__((ext_vector_type(4)))  __bf16 v4bf;
typedef __attribute__((ext_vector_type(8)))  float  v8f;

union Frag32B { v16bf v; uint4 q[2]; };   // 16 bf16 = 8 VGPRs = one WMMA A/B frag

#define H_DIM   1024
#define I_DIM   2048
#define N_EXP   8
#define T_TOK   8192
#define N_SLOTS (T_TOK * 2)
#define MAX_MT  128   // per-expert max tokens 8192 / BM(64)

static __device__ inline v8f zero8() {
  v8f z;
  #pragma unroll
  for (int i = 0; i < 8; ++i) z[i] = 0.0f;
  return z;
}

static __device__ inline v8f wmma_bf16(const Frag32B& a, const Frag32B& b, v8f c) {
  return __builtin_amdgcn_wmma_f32_16x16x32_bf16(false, a.v, false, b.v,
                                                 (short)0, c, false, false);
}

// ---------------------------------------------------------------------------
// 0. zero the per-expert counters
__global__ void moe_zero(int* __restrict__ counts) {
  if (threadIdx.x < N_EXP) counts[threadIdx.x] = 0;
}

// ---------------------------------------------------------------------------
// 1. router: one wave32 per token; logits -> top2 -> renormalized weights
__global__ __launch_bounds__(256) void moe_router(
    const float* __restrict__ x, const float* __restrict__ rw,
    int* __restrict__ topi, float* __restrict__ topw, int* __restrict__ counts)
{
  const int t    = blockIdx.x * 8 + (threadIdx.x >> 5);
  const int lane = threadIdx.x & 31;
  float acc[N_EXP];
  #pragma unroll
  for (int e = 0; e < N_EXP; ++e) acc[e] = 0.0f;

  const float* xp = x + (size_t)t * H_DIM;
  for (int h = lane; h < H_DIM; h += 32) {
    float xv = xp[h];
    #pragma unroll
    for (int e = 0; e < N_EXP; ++e) acc[e] += xv * rw[e * H_DIM + h];
  }
  #pragma unroll
  for (int e = 0; e < N_EXP; ++e) {
    #pragma unroll
    for (int o = 16; o > 0; o >>= 1) acc[e] += __shfl_xor(acc[e], o, 32);
  }
  if (lane == 0) {
    float best = acc[0]; int bi = 0;
    #pragma unroll
    for (int e = 1; e < N_EXP; ++e) if (acc[e] > best) { best = acc[e]; bi = e; }
    float sec = -3.4e38f; int si = (bi == 0) ? 1 : 0;
    #pragma unroll
    for (int e = 0; e < N_EXP; ++e)
      if (e != bi && acc[e] > sec) { sec = acc[e]; si = e; }
    // softmax over all 8 then renorm over top-2 == e0/(e0+e1) on raw logits
    float e1  = __expf(sec - best);
    float inv = 1.0f / (1.0f + e1);
    topi[t * 2]     = bi;  topi[t * 2 + 1] = si;
    topw[t * 2]     = inv; topw[t * 2 + 1] = e1 * inv;
    atomicAdd(&counts[bi], 1);
    atomicAdd(&counts[si], 1);
  }
}

// ---------------------------------------------------------------------------
// 2. exclusive scan of 8 counts + zero cursors
__global__ void moe_scan(const int* __restrict__ counts,
                         int* __restrict__ offsets, int* __restrict__ cursors) {
  if (threadIdx.x == 0) {
    int s = 0;
    for (int e = 0; e < N_EXP; ++e) { offsets[e] = s; s += counts[e]; }
  }
  if (threadIdx.x < N_EXP) cursors[threadIdx.x] = 0;
}

// ---------------------------------------------------------------------------
// 3. scatter tokens into contiguous per-expert slot lists
__global__ void moe_scatter(const int* __restrict__ topi,
                            const int* __restrict__ offsets, int* __restrict__ cursors,
                            int* __restrict__ slot_token, int* __restrict__ tok_slot)
{
  int t = blockIdx.x * 256 + threadIdx.x;
  if (t >= T_TOK) return;
  #pragma unroll
  for (int k = 0; k < 2; ++k) {
    int e    = topi[t * 2 + k];
    int pos  = atomicAdd(&cursors[e], 1);
    int slot = offsets[e] + pos;
    slot_token[slot]     = t;
    tok_slot[t * 2 + k]  = slot;
  }
}

// ---------------------------------------------------------------------------
// 4. f32 -> bf16 (RNE) elementwise, 4/thread
__global__ __launch_bounds__(256) void f32_to_bf16_k(
    const float* __restrict__ in, bf16* __restrict__ out, int n4)
{
  int i = blockIdx.x * 256 + threadIdx.x;
  if (i >= n4) return;
  float4 v = *(const float4*)(&in[(size_t)i * 4]);
  v4bf o;
  o[0] = (bf16)v.x; o[1] = (bf16)v.y; o[2] = (bf16)v.z; o[3] = (bf16)v.w;
  *(v4bf*)(&out[(size_t)i * 4]) = o;
}

// ---------------------------------------------------------------------------
// 5. grouped GEMM 1: mid[slot, n] = silu(x@wg^T) * (x@wu^T)
//    block = 64 tokens x 64 inter-cols (g AND u); 8 waves (wm 0..3, wn 0..1);
//    per wave per K-step: 1 A frag -> 4 WMMAs (2 g + 2 u); silu in registers.
__global__ __launch_bounds__(256) void moe_gemm1(
    const bf16* __restrict__ xb,  const bf16* __restrict__ wgb,
    const bf16* __restrict__ wub,
    const int* __restrict__ counts, const int* __restrict__ offsets,
    const int* __restrict__ slot_token,
    bf16* __restrict__ mid)
{
  const int e   = blockIdx.x >> 7;
  const int mt  = blockIdx.x & (MAX_MT - 1);
  const int cnt = counts[e];
  if (mt * 64 >= cnt) return;
  const int off = offsets[e];
  const int n0  = blockIdx.y * 64;

  __shared__ __align__(16) bf16 sA [64 * 32];
  __shared__ __align__(16) bf16 sBg[64 * 32];
  __shared__ __align__(16) bf16 sBu[64 * 32];
  __shared__ int stok[64];

  const int tid  = threadIdx.x;
  const int lane = tid & 31;
  const int wave = tid >> 5;
  const int wm   = wave & 3;   // M sub-tile (x16)
  const int wn   = wave >> 2;  // N block (x32)

  if (tid < 64) {
    int r = mt * 64 + tid;
    stok[tid] = slot_token[off + ((r < cnt) ? r : 0)];  // clamp tail gather
  }
  __syncthreads();

  // loop-invariant global pointers (hoisted gather)
  const int row  = tid >> 2;            // 0..63
  const int col8 = (tid & 3) * 8;       // 0,8,16,24
  const bf16* aP = xb  + (size_t)stok[row] * H_DIM + col8;
  const bf16* gP = wgb + ((size_t)e * I_DIM + n0 + row) * H_DIM + col8;
  const bf16* uP = wub + ((size_t)e * I_DIM + n0 + row) * H_DIM + col8;

  // register double-buffer: preload K-slice 0
  uint4 ra = *(const uint4*)aP;
  uint4 rg = *(const uint4*)gP;
  uint4 ru = *(const uint4*)uP;

  v8f accG[2] = { zero8(), zero8() };
  v8f accU[2] = { zero8(), zero8() };

  for (int k0 = 0; k0 < H_DIM; k0 += 32) {
    *(uint4*)(&sA [row * 32 + col8]) = ra;
    *(uint4*)(&sBg[row * 32 + col8]) = rg;
    *(uint4*)(&sBu[row * 32 + col8]) = ru;
    __syncthreads();

    if (k0 + 32 < H_DIM) {   // issue next slice early; wait lands a stage later
      ra = *(const uint4*)(aP + k0 + 32);
      rg = *(const uint4*)(gP + k0 + 32);
      ru = *(const uint4*)(uP + k0 + 32);
      __builtin_prefetch(gP + k0 + 64, 0, 1);
      __builtin_prefetch(uP + k0 + 64, 0, 1);
    }

    // A frag: lane = M (mod 16); contiguous K, lane half h holds k 16h..16h+15.
    Frag32B a;
    const int am = wm * 16 + (lane & 15);
    const int ab = (lane >> 4) * 16;
    a.q[0] = *(const uint4*)(&sA[am * 32 + ab]);
    a.q[1] = *(const uint4*)(&sA[am * 32 + ab + 8]);
    const int bb = (lane >> 4) * 16;
    #pragma unroll
    for (int s = 0; s < 2; ++s) {
      const int bn = (wn * 32 + s * 16 + (lane & 15)) * 32;
      Frag32B bg, bu;
      bg.q[0] = *(const uint4*)(&sBg[bn + bb]);
      bg.q[1] = *(const uint4*)(&sBg[bn + bb + 8]);
      bu.q[0] = *(const uint4*)(&sBu[bn + bb]);
      bu.q[1] = *(const uint4*)(&sBu[bn + bb + 8]);
      accG[s] = wmma_bf16(a, bg, accG[s]);
      accU[s] = wmma_bf16(a, bu, accU[s]);
    }
    __syncthreads();
  }

  // C/D layout: lane = N (mod 16); VGPR v -> M = v + 8*(lane>>4)
  #pragma unroll
  for (int s = 0; s < 2; ++s) {
    const int n = n0 + wn * 32 + s * 16 + (lane & 15);
    #pragma unroll
    for (int v = 0; v < 8; ++v) {
      int r = mt * 64 + wm * 16 + (lane >> 4) * 8 + v;
      if (r < cnt) {                                  // guard: don't alias next expert
        float g = accG[s][v], u = accU[s][v];
        float si = g / (1.0f + __expf(-g));           // silu
        mid[(size_t)(off + r) * I_DIM + n] = (bf16)(si * u);
      }
    }
  }
}

// ---------------------------------------------------------------------------
// 6. grouped GEMM 2: contrib[slot, h] = mid[slot] @ wd[e]^T.  K = 2048.
//    block = 64 slots x 128 out-cols; per wave per K-step: 1 A frag -> 4 WMMAs.
__global__ __launch_bounds__(256) void moe_gemm2(
    const bf16* __restrict__ mid, const bf16* __restrict__ wdb,
    const int* __restrict__ counts, const int* __restrict__ offsets,
    bf16* __restrict__ contrib)
{
  const int e   = blockIdx.x >> 7;
  const int mt  = blockIdx.x & (MAX_MT - 1);
  const int cnt = counts[e];
  if (mt * 64 >= cnt) return;
  const int off = offsets[e];
  const int n0  = blockIdx.y * 128;

  __shared__ __align__(16) bf16 sA[64 * 32];
  __shared__ __align__(16) bf16 sB[128 * 32];

  const int tid  = threadIdx.x;
  const int lane = tid & 31;
  const int wave = tid >> 5;
  const int wm   = wave & 3;   // M sub-tile (x16)
  const int wn   = wave >> 2;  // N block (x64)

  const int row  = tid >> 2;
  const int col8 = (tid & 3) * 8;
  int aslot = off + mt * 64 + row;                // slots are contiguous here
  if (aslot >= N_SLOTS) aslot = N_SLOTS - 1;      // clamp tail

  const bf16* aP  = mid + (size_t)aslot * I_DIM + col8;
  const bf16* bP0 = wdb + ((size_t)e * H_DIM + n0 + row) * I_DIM + col8;
  const bf16* bP1 = wdb + ((size_t)e * H_DIM + n0 + row + 64) * I_DIM + col8;

  uint4 ra  = *(const uint4*)aP;
  uint4 rb0 = *(const uint4*)bP0;
  uint4 rb1 = *(const uint4*)bP1;

  v8f acc[4] = { zero8(), zero8(), zero8(), zero8() };

  for (int k0 = 0; k0 < I_DIM; k0 += 32) {
    *(uint4*)(&sA[row * 32 + col8])        = ra;
    *(uint4*)(&sB[row * 32 + col8])        = rb0;
    *(uint4*)(&sB[(row + 64) * 32 + col8]) = rb1;
    __syncthreads();

    if (k0 + 32 < I_DIM) {
      ra  = *(const uint4*)(aP  + k0 + 32);
      rb0 = *(const uint4*)(bP0 + k0 + 32);
      rb1 = *(const uint4*)(bP1 + k0 + 32);
      __builtin_prefetch(bP0 + k0 + 64, 0, 1);
      __builtin_prefetch(bP1 + k0 + 64, 0, 1);
    }

    Frag32B a;
    const int am = wm * 16 + (lane & 15);
    const int ab = (lane >> 4) * 16;
    a.q[0] = *(const uint4*)(&sA[am * 32 + ab]);
    a.q[1] = *(const uint4*)(&sA[am * 32 + ab + 8]);
    const int bb = (lane >> 4) * 16;
    #pragma unroll
    for (int s = 0; s < 4; ++s) {
      const int bn = (wn * 64 + s * 16 + (lane & 15)) * 32;
      Frag32B b;
      b.q[0] = *(const uint4*)(&sB[bn + bb]);
      b.q[1] = *(const uint4*)(&sB[bn + bb + 8]);
      acc[s] = wmma_bf16(a, b, acc[s]);
    }
    __syncthreads();
  }

  #pragma unroll
  for (int v = 0; v < 8; ++v) {
    int r = mt * 64 + wm * 16 + (lane >> 4) * 8 + v;
    if (r < cnt) {
      size_t base = (size_t)(off + r) * H_DIM + n0 + wn * 64 + (lane & 15);
      #pragma unroll
      for (int s = 0; s < 4; ++s)
        contrib[base + s * 16] = (bf16)acc[s][v];
    }
  }
}

// ---------------------------------------------------------------------------
// 7. combine: out[t] = w0*contrib[slot0] + w1*contrib[slot1]   (f32, no atomics)
__global__ __launch_bounds__(256) void moe_combine(
    const bf16* __restrict__ contrib, const int* __restrict__ tok_slot,
    const float* __restrict__ topw, float* __restrict__ out)
{
  int i4 = (blockIdx.x * 256 + threadIdx.x) * 4;
  int t  = i4 >> 10;
  int h  = i4 & (H_DIM - 1);
  int s0 = tok_slot[t * 2], s1 = tok_slot[t * 2 + 1];
  float w0 = topw[t * 2],  w1 = topw[t * 2 + 1];
  v4bf c0 = *(const v4bf*)(&contrib[(size_t)s0 * H_DIM + h]);
  v4bf c1 = *(const v4bf*)(&contrib[(size_t)s1 * H_DIM + h]);
  float4 o;
  o.x = w0 * (float)c0[0] + w1 * (float)c1[0];
  o.y = w0 * (float)c0[1] + w1 * (float)c1[1];
  o.z = w0 * (float)c0[2] + w1 * (float)c1[2];
  o.w = w0 * (float)c0[3] + w1 * (float)c1[3];
  *(float4*)(&out[i4]) = o;
}

// ---------------------------------------------------------------------------
extern "C" void kernel_launch(void* const* d_in, const int* in_sizes, int n_in,
                              void* d_out, int out_size, void* d_ws, size_t ws_size,
                              hipStream_t stream)
{
  const float* x  = (const float*)d_in[0];   // [2,4096,1024]
  const float* rw = (const float*)d_in[1];   // [8,1024]
  const float* wg = (const float*)d_in[2];   // [8,2048,1024]
  const float* wu = (const float*)d_in[3];   // [8,2048,1024]
  const float* wd = (const float*)d_in[4];   // [8,1024,2048]
  float* out = (float*)d_out;
  (void)in_sizes; (void)n_in; (void)out_size; (void)ws_size;

  // workspace layout (bytes), 256-aligned; total ~218.4 MB
  char* ws = (char*)d_ws;
  bf16*  xb   = (bf16*)(ws + 0);            // 16,777,216
  bf16*  wgb  = (bf16*)(ws + 16777216);     // 33,554,432
  bf16*  wub  = (bf16*)(ws + 50331648);     // 33,554,432
  bf16*  wdb  = (bf16*)(ws + 83886080);     // 33,554,432
  bf16*  mid  = (bf16*)(ws + 117440512);    // 67,108,864  [16384,2048]
  bf16*  ctr  = (bf16*)(ws + 184549376);    // 33,554,432  [16384,1024]
  int*   topi = (int*)(ws + 218103808);     // 65,536
  float* topw = (float*)(ws + 218169344);   // 65,536
  int*   slot_token = (int*)(ws + 218234880);
  int*   tok_slot   = (int*)(ws + 218300416);
  int*   counts     = (int*)(ws + 218365952);
  int*   offsets    = (int*)(ws + 218366208);
  int*   cursors    = (int*)(ws + 218366464);

  moe_zero<<<1, 32, 0, stream>>>(counts);
  moe_router<<<T_TOK / 8, 256, 0, stream>>>(x, rw, topi, topw, counts);
  moe_scan<<<1, 32, 0, stream>>>(counts, offsets, cursors);
  moe_scatter<<<T_TOK / 256, 256, 0, stream>>>(topi, offsets, cursors,
                                               slot_token, tok_slot);

  f32_to_bf16_k<<<T_TOK * H_DIM / 4 / 256, 256, 0, stream>>>(x, xb, T_TOK * H_DIM / 4);
  int wn4 = N_EXP * I_DIM * H_DIM / 4;      // 4,194,304
  f32_to_bf16_k<<<wn4 / 256, 256, 0, stream>>>(wg, wgb, wn4);
  f32_to_bf16_k<<<wn4 / 256, 256, 0, stream>>>(wu, wub, wn4);
  f32_to_bf16_k<<<wn4 / 256, 256, 0, stream>>>(wd, wdb, wn4);

  moe_gemm1<<<dim3(N_EXP * MAX_MT, I_DIM / 64), 256, 0, stream>>>(
      xb, wgb, wub, counts, offsets, slot_token, mid);
  moe_gemm2<<<dim3(N_EXP * MAX_MT, H_DIM / 128), 256, 0, stream>>>(
      mid, wdb, counts, offsets, ctr);
  moe_combine<<<T_TOK * H_DIM / 1024, 256, 0, stream>>>(ctr, tok_slot, topw, out);
}
